// SelfAttentionBlock2D_42417097016577
// MI455X (gfx1250) — matrix operand order repaired and checked by reading, not compiled, not run on hardware
//
#include <hip/hip_runtime.h>
#include <hip/hip_bf16.h>

// MI455X (gfx1250) self-attention block, bf16 WMMA (v_wmma_f32_16x16x32_bf16),
// f32 accumulation. 3 kernels: projections -> LDS-resident attention -> out proj.

#define B_      4
#define C_IN_   512
#define N_      4096
#define K_CH_   256
#define V_CH_   256
#define OUT_CH_ 512
#define BN_EPS_ 1e-5f
#define SCALE_  0.0625f  // 256^-0.5

typedef __attribute__((ext_vector_type(16))) __bf16          v16bf;
typedef __attribute__((ext_vector_type(16))) unsigned short  v16u;
typedef __attribute__((ext_vector_type(8)))  float           v8f;

static __device__ __forceinline__ unsigned short f2bf(float f) {
  __bf16 h = (__bf16)f;                       // hardware cvt path
  return __builtin_bit_cast(unsigned short, h);
}

static __device__ __forceinline__ v8f wmma_bf16(v16bf a, v16bf b, v8f c) {
  return __builtin_amdgcn_wmma_f32_16x16x32_bf16(false, a, false, b,
                                                 (short)0, c, false, false);
}

// A-fragment (16x32 bf16) K index for element j:  j + 16*half, row = lane&15
// B-fragment (32x16 bf16) K index for element j:  j + 16*half, col = lane&15
// C/D f32 tile:                                   row = r + 8*half, col = lane&15

// ---------------------------------------------------------------------------
// Kernel 1: fused key/query + value projections.
//   D[o, n] = sum_c W[o,c] * x[b,c,n]
//   key path (o<256): +key_b, BN (inference), ReLU -> qk[b][n][k]  (bf16)
//   val path        : +value_b                    -> vws[b][v][n]  (bf16)
// Each wave: one 16-channel o-tile x 8 n-tiles (128 pixels); weight A-fragment
// is built once per K-chunk and reused across all 8 n-tiles.
// ---------------------------------------------------------------------------
__global__ __launch_bounds__(256) void proj_kernel(
    const float* __restrict__ x, const float* __restrict__ key_w,
    const float* __restrict__ key_b, const float* __restrict__ bn_gamma,
    const float* __restrict__ bn_beta, const float* __restrict__ bn_mean,
    const float* __restrict__ bn_var, const float* __restrict__ value_w,
    const float* __restrict__ value_b, unsigned short* __restrict__ qk,
    unsigned short* __restrict__ vws) {
  const int wave = threadIdx.x >> 5;
  const int lane = threadIdx.x & 31;
  const int half = lane >> 4;
  const int l16  = lane & 15;
  const int tile   = blockIdx.x * 8 + wave;     // 4096 wave-tiles
  const int o_tile = tile & 31;                 // 32 tiles of 16 channels
  const int nblk   = (tile >> 5) & 31;          // 32 blocks of 128 pixels
  const int b      = tile >> 10;
  const int nbase  = nblk * 128;

  const bool   is_key = (o_tile < 16);
  const float* wsrc   = is_key ? key_w : value_w;
  const int    orow   = (is_key ? o_tile : (o_tile - 16)) * 16 + l16;

  v8f acc[8];
#pragma unroll
  for (int nt = 0; nt < 8; ++nt) acc[nt] = (v8f){};

  for (int cc = 0; cc < 16; ++cc) {             // K loop: 512 / 32
    const int c0 = cc * 32;
    v16bf a;
#pragma unroll
    for (int j = 0; j < 16; ++j) {              // weight A-fragment, built once
      const int ca = c0 + j + 16 * half;
      a[j] = (__bf16)wsrc[(size_t)orow * C_IN_ + ca];
    }
#pragma unroll
    for (int nt = 0; nt < 8; ++nt) {            // reuse A across 8 n-tiles
      const int nb = nbase + nt * 16;
      v16bf bfr;
#pragma unroll
      for (int j = 0; j < 16; ++j) {
        const int cb = c0 + j + 16 * half;
        bfr[j] = (__bf16)x[((size_t)b * C_IN_ + cb) * N_ + nb + l16];
      }
      acc[nt] = wmma_bf16(a, bfr, acc[nt]);
    }
  }

  if (is_key) {
    float inv[8], mu[8], beta[8], kb[8];
#pragma unroll
    for (int r = 0; r < 8; ++r) {
      const int k = o_tile * 16 + r + 8 * half;
      inv[r]  = bn_gamma[k] * rsqrtf(bn_var[k] + BN_EPS_);
      mu[r]   = bn_mean[k];
      beta[r] = bn_beta[k];
      kb[r]   = key_b[k];
    }
#pragma unroll
    for (int nt = 0; nt < 8; ++nt) {
      const int nb = nbase + nt * 16;
#pragma unroll
      for (int r = 0; r < 8; ++r) {
        const int k = o_tile * 16 + r + 8 * half;
        float v = (acc[nt][r] + kb[r] - mu[r]) * inv[r] + beta[r];
        v = fmaxf(v, 0.0f);
        qk[((size_t)b * N_ + nb + l16) * K_CH_ + k] = f2bf(v);
      }
    }
  } else {
    float vb[8];
#pragma unroll
    for (int r = 0; r < 8; ++r)
      vb[r] = value_b[(o_tile - 16) * 16 + r + 8 * half];
#pragma unroll
    for (int nt = 0; nt < 8; ++nt) {
      const int nb = nbase + nt * 16;
#pragma unroll
      for (int r = 0; r < 8; ++r) {
        const int vch = (o_tile - 16) * 16 + r + 8 * half;
        vws[((size_t)b * V_CH_ + vch) * N_ + nb + l16] = f2bf(acc[nt][r] + vb[r]);
      }
    }
  }
}

// ---------------------------------------------------------------------------
// Kernel 2: attention core. One workgroup (4 waves) per (batch, 16-row block).
// 16x4096 f32 score slab lives in LDS (256 KB of the 320 KB WGP budget).
//   phase 1: S = SCALE * Q K^T   (WMMA, Q fragments register-resident)
//   phase 2: row max, exp in place (f32), row sums
//   phase 3: O += P V via WMMA; P A-fragments are the HIGH u16 of the f32 exp
//            values in LDS (bf16 truncation) -> zero conversion cost
//   phase 4: cross-wave ds_add_f32 reduction, normalize, bf16 ctx out
// ---------------------------------------------------------------------------
__global__ __launch_bounds__(128) void attn_kernel(
    const unsigned short* __restrict__ qk,
    const unsigned short* __restrict__ vws,
    unsigned short* __restrict__ ctx) {
  extern __shared__ char smem_raw[];
  float* S      = (float*)smem_raw;        // 16 x 4096
  float* Obuf   = S + 16 * N_;             // 16 x 256 (also stat scratch)
  float* rowmax = Obuf + 16 * V_CH_;       // 16
  float* rowsum = rowmax + 16;             // 16

  const int wave = threadIdx.x >> 5;
  const int lane = threadIdx.x & 31;
  const int half = lane >> 4;
  const int l16  = lane & 15;
  const int b     = blockIdx.x >> 8;
  const int nbase = (blockIdx.x & 255) * 16;

  // Q fragments: 16 rows x 256, held in registers for the whole kernel.
  v16bf qa[8];
#pragma unroll
  for (int kc = 0; kc < 8; ++kc) {
    v16u u;
#pragma unroll
    for (int j = 0; j < 16; ++j) {
      const int k = kc * 32 + j + 16 * half;
      u[j] = qk[((size_t)b * N_ + nbase + l16) * K_CH_ + k];
    }
    qa[kc] = __builtin_bit_cast(v16bf, u);
  }

  // Phase 1: each wave owns a 1024-column m-slice of S.
  const int mslice = wave * 1024;
  for (int mt = 0; mt < 64; ++mt) {
    const int mb = mslice + mt * 16;
    const unsigned short* kp =
        qk + ((size_t)b * N_ + mb + l16) * K_CH_ + 16 * half;
    v8f acc = {};
#pragma unroll
    for (int kc = 0; kc < 8; ++kc) {
      v16u u = *reinterpret_cast<const v16u*>(kp + kc * 32);  // 32B contiguous
      acc = wmma_bf16(qa[kc], __builtin_bit_cast(v16bf, u), acc);
    }
#pragma unroll
    for (int r = 0; r < 8; ++r)
      S[(r + 8 * half) * N_ + mb + l16] = acc[r] * SCALE_;
  }
  __syncthreads();

  // Phase 2: softmax statistics. 8 threads per row, 512 columns each.
  const int tid = threadIdx.x;
  const int row = tid >> 3;
  const int seg = tid & 7;
  {
    float mx = -1e30f;
    const float* Sr = S + row * N_ + seg * 512;
    for (int c = 0; c < 512; ++c) mx = fmaxf(mx, Sr[c]);
    Obuf[tid] = mx;
  }
  __syncthreads();
  if (seg == 0) {
    float m = Obuf[row * 8];
    for (int i = 1; i < 8; ++i) m = fmaxf(m, Obuf[row * 8 + i]);
    rowmax[row] = m;
  }
  __syncthreads();
  {
    const float rm = rowmax[row];
    float sum = 0.0f;
    float* Sr = S + row * N_ + seg * 512;
    for (int c = 0; c < 512; ++c) {
      const float e = __expf(Sr[c] - rm);
      Sr[c] = e;                            // exp'd P stays f32 in place
      sum += e;
    }
    Obuf[tid] = sum;
  }
  __syncthreads();
  if (seg == 0) {
    float s = 0.0f;
    for (int i = 0; i < 8; ++i) s += Obuf[row * 8 + i];
    rowsum[row] = s;
  }
  __syncthreads();
  for (int i = tid; i < 16 * V_CH_; i += 128) Obuf[i] = 0.0f;
  __syncthreads();

  // Phase 3: O = P V over this wave's m-slice. 16 D-tiles cover v = 0..255.
  const unsigned short* Su = reinterpret_cast<const unsigned short*>(S);
  v8f oacc[16];
#pragma unroll
  for (int t = 0; t < 16; ++t) oacc[t] = (v8f){};
  for (int mc = 0; mc < 32; ++mc) {
    const int mb = mslice + mc * 32;
    v16u pu;
#pragma unroll
    for (int j = 0; j < 16; ++j) {          // high u16 of f32 == bf16 (trunc)
      const int m = mb + j + 16 * half;
      pu[j] = Su[2 * (l16 * N_ + m) + 1];
    }
    const v16bf pa = __builtin_bit_cast(v16bf, pu);
    const unsigned short* vp = vws + (size_t)b * V_CH_ * N_ + mb + 16 * half;
#pragma unroll
    for (int t = 0; t < 16; ++t) {
      v16u u = *reinterpret_cast<const v16u*>(vp + (size_t)(t * 16 + l16) * N_);
      oacc[t] = wmma_bf16(pa, __builtin_bit_cast(v16bf, u), oacc[t]);
    }
  }
  // Cross-wave reduction into shared O buffer.
#pragma unroll
  for (int t = 0; t < 16; ++t)
#pragma unroll
    for (int r = 0; r < 8; ++r)
      atomicAdd(&Obuf[(r + 8 * half) * V_CH_ + t * 16 + l16], oacc[t][r]);
  __syncthreads();

  // Phase 4: normalize and emit bf16 ctx[b][n][v].
  for (int i = tid; i < 16 * V_CH_; i += 128) {
    const int rr = i >> 8;                  // / V_CH_
    const float v = Obuf[i] / rowsum[rr];
    ctx[((size_t)b * N_ + nbase + rr) * V_CH_ + (i & 255)] = f2bf(v);
  }
}

// ---------------------------------------------------------------------------
// Kernel 3: output projection out[b][o][n] = w_w[o,:] . ctx[b][n,:] + w_b[o]
// Each wave: one o-tile x 8 n-tiles; weight A-fragment reused across n-tiles.
// ---------------------------------------------------------------------------
__global__ __launch_bounds__(256) void outproj_kernel(
    const unsigned short* __restrict__ ctx, const float* __restrict__ w_w,
    const float* __restrict__ w_b, float* __restrict__ out) {
  const int wave = threadIdx.x >> 5;
  const int lane = threadIdx.x & 31;
  const int half = lane >> 4;
  const int l16  = lane & 15;
  const int tile   = blockIdx.x * 8 + wave;     // 4096 wave-tiles
  const int o_tile = tile & 31;
  const int nblk   = (tile >> 5) & 31;
  const int b      = tile >> 10;
  const int orow   = o_tile * 16 + l16;
  const int nbase  = nblk * 128;

  v8f acc[8];
#pragma unroll
  for (int nt = 0; nt < 8; ++nt) acc[nt] = (v8f){};

#pragma unroll
  for (int vc = 0; vc < 8; ++vc) {              // K loop: 256 / 32
    v16bf a;
#pragma unroll
    for (int j = 0; j < 16; ++j) {
      const int v = vc * 32 + j + 16 * half;
      a[j] = (__bf16)w_w[(size_t)orow * V_CH_ + v];
    }
#pragma unroll
    for (int nt = 0; nt < 8; ++nt) {
      const unsigned short* cp =
          ctx + ((size_t)b * N_ + nbase + nt * 16 + l16) * V_CH_ + 16 * half;
      v16u bu = *reinterpret_cast<const v16u*>(cp + vc * 32);  // 32B contiguous
      acc[nt] = wmma_bf16(a, __builtin_bit_cast(v16bf, bu), acc[nt]);
    }
  }

  float wb[8];
#pragma unroll
  for (int r = 0; r < 8; ++r) wb[r] = w_b[o_tile * 16 + r + 8 * half];
#pragma unroll
  for (int nt = 0; nt < 8; ++nt) {
    const int nb = nbase + nt * 16;
#pragma unroll
    for (int r = 0; r < 8; ++r) {
      const int o = o_tile * 16 + r + 8 * half;
      out[((size_t)b * OUT_CH_ + o) * N_ + nb + l16] = acc[nt][r] + wb[r];
    }
  }
}

extern "C" void kernel_launch(void* const* d_in, const int* in_sizes, int n_in,
                              void* d_out, int out_size, void* d_ws,
                              size_t ws_size, hipStream_t stream) {
  const float* x        = (const float*)d_in[0];
  const float* key_w    = (const float*)d_in[1];
  const float* key_b    = (const float*)d_in[2];
  const float* bn_gamma = (const float*)d_in[3];
  const float* bn_beta  = (const float*)d_in[4];
  const float* bn_mean  = (const float*)d_in[5];
  const float* bn_var   = (const float*)d_in[6];
  const float* value_w  = (const float*)d_in[7];
  const float* value_b  = (const float*)d_in[8];
  const float* w_w      = (const float*)d_in[9];
  const float* w_b      = (const float*)d_in[10];
  float* out = (float*)d_out;

  // Workspace: bf16 qk[b][n][k], v[b][v][n], ctx[b][n][v]  (8 MB each).
  unsigned short* qk  = (unsigned short*)d_ws;
  unsigned short* vws = qk + (size_t)B_ * N_ * K_CH_;
  unsigned short* ctx = vws + (size_t)B_ * V_CH_ * N_;

  proj_kernel<<<512, 256, 0, stream>>>(x, key_w, key_b, bn_gamma, bn_beta,
                                       bn_mean, bn_var, value_w, value_b, qk,
                                       vws);

  const int smem = (16 * N_ + 16 * V_CH_ + 32) * (int)sizeof(float);  // ~272 KB
  (void)hipFuncSetAttribute(reinterpret_cast<const void*>(attn_kernel),
                            hipFuncAttributeMaxDynamicSharedMemorySize, smem);
  attn_kernel<<<B_ * 256, 128, smem, stream>>>(qk, vws, ctx);

  outproj_kernel<<<512, 256, 0, stream>>>(ctx, w_w, w_b, out);
}
